// VAE_conv_ENCODER_3_19464791786171
// MI455X (gfx1250) — matrix operand order repaired and checked by reading, not compiled, not run on hardware
//
#include <hip/hip_runtime.h>
#include <hip/hip_bf16.h>
#include <math.h>

// ---------------------------------------------------------------------------
// GNN-VAE encoder for MI455X (gfx1250, wave32).
//  * Dense projections x@W^T+b as f16 WMMA (v_wmma_f32_16x16x32_f16), f32
//    accumulate. GEMM templated on <M,K> (only 3 shapes exist) so the K-step
//    and column-tile loops are fully unrolled straight-line WMMA chains; LDS
//    footprint is exact per shape. 64-row blocks amortize the W LDS stage;
//    each wave holds its A fragments in registers across its column tiles.
//    Stores use a block-uniform fast path (no exec-masked per-element guards).
//  * Attention aggregation is gather-based via a dst-CSR built once per call:
//    wave-per-node segment max/softmax/weighted-sum in registers + wave32
//    butterflies. NO atomics in the per-layer path. All node tensors
//    (<=25.6MB) sit in the 192MB L2; gathers are L2 traffic, far below the
//    23.3TB/s HBM roof.
//  * Set2Set uses sorted batch_index: graph-CSR ranges, one fused
//    wave-per-graph kernel per step -> no atomics.
//  * Heads reuse the WMMA GEMM; reparameterization uses PCG/Box-Muller
//    (jax threefry noise not reproducible; compile-only loop).
// ---------------------------------------------------------------------------

typedef __attribute__((ext_vector_type(16))) _Float16 v16h;
typedef __attribute__((ext_vector_type(8)))  float    v8f;

#define N_NODES  50000
#define N_EDGES  500000
#define N_GRAPHS 2048
#define H_SIZE   64
#define EMB      128
#define ENC      64
#define S2S_STEPS 4
#define NUM_LAYERS 4
#define TROWS    64            // GEMM rows per block

// ---------------------------------------------------------------- utilities
__global__ void fill_f32(float* __restrict__ p, float v, int n) {
  int i = blockIdx.x * blockDim.x + threadIdx.x;
  if (i < n) p[i] = v;
}
__global__ void fill_i32(int* __restrict__ p, int v, int n) {
  int i = blockIdx.x * blockDim.x + threadIdx.x;
  if (i < n) p[i] = v;
}
__global__ void copy_i32(const int* __restrict__ a, int* __restrict__ b, int n) {
  int i = blockIdx.x * blockDim.x + threadIdx.x;
  if (i < n) b[i] = a[i];
}

__device__ __forceinline__ float waveSum(float v) {
#pragma unroll
  for (int off = 16; off; off >>= 1) v += __shfl_xor(v, off, 32);
  return v;  // full value in all 32 lanes (butterfly)
}
__device__ __forceinline__ float waveMax(float v) {
#pragma unroll
  for (int off = 16; off; off >>= 1) v = fmaxf(v, __shfl_xor(v, off, 32));
  return v;
}

// ------------------------------------------------------------- CSR building
__global__ void count_kernel(const int* __restrict__ idx, int* __restrict__ cnt, int n) {
  int i = blockIdx.x * blockDim.x + threadIdx.x;
  if (i < n) atomicAdd(&cnt[idx[i]], 1);
}

// exclusive prefix scan, single block of 1024 threads (run once per call)
__global__ void scan_kernel(const int* __restrict__ cnt, int* __restrict__ ptr, int n) {
  __shared__ int part[1024];
  int tid = threadIdx.x;
  int chunk = (n + 1023) >> 10;
  int base = tid * chunk;
  int s = 0;
  for (int j = 0; j < chunk; ++j) { int k = base + j; if (k < n) s += cnt[k]; }
  part[tid] = s;
  __syncthreads();
  for (int off = 1; off < 1024; off <<= 1) {
    int v = (tid >= off) ? part[tid - off] : 0;
    __syncthreads();
    part[tid] += v;
    __syncthreads();
  }
  int run = (tid == 0) ? 0 : part[tid - 1];
  for (int j = 0; j < chunk; ++j) {
    int k = base + j;
    if (k < n) { ptr[k] = run; run += cnt[k]; }
  }
  if (tid == 0) ptr[n] = part[1023];
}

__global__ void bucket_kernel(const int* __restrict__ dstv, int* __restrict__ cursor,
                              int* __restrict__ eid, int E) {
  int e = blockIdx.x * blockDim.x + threadIdx.x;
  if (e >= E) return;
  int pos = atomicAdd(&cursor[dstv[e]], 1);
  eid[pos] = e;
}

// ---------------------------------------------------------------- WMMA GEMM
// Out[N,M] = X[N,K] @ W[M,K]^T + b[M]   (f32 in/out, f16 multiply, f32 acc)
// Shapes are compile-time: straight-line WMMA chains, exact LDS footprint.
// grid.x = ceil(N/64); block = 256 = 8 wave32s. Wave w: row tile rt = w&3,
// column group w>>2; A fragments live in registers across the column tiles.
template <int M, int K>
__global__ void gemm_bias_wmma(const float* __restrict__ X,
                               const float* __restrict__ W,
                               const float* __restrict__ b,
                               float* __restrict__ Out, int N) {
  constexpr int NK  = K / 32;        // k-steps (2 or 4)
  constexpr int PER = (M / 16) / 2;  // column tiles per wave (2 or 4)
  __shared__ _Float16 sW[M * K];
  __shared__ _Float16 sX[TROWS * K];
  const int tid  = threadIdx.x;
  const int row0 = blockIdx.x * TROWS;

  for (int i = tid; i < M * K; i += 256) sW[i] = (_Float16)W[i];
  for (int i = tid; i < TROWS * K; i += 256) {
    int r = i / K, c = i - r * K;
    int gr = row0 + r;
    sX[i] = (gr < N) ? (_Float16)X[(size_t)gr * K + c] : (_Float16)0.0f;
  }
  __syncthreads();

  const int wave = tid >> 5;
  const int lane = tid & 31;
  const int rt   = wave & 3;           // 16-row tile within the 64-row block
  const int cgrp = wave >> 2;          // 0..1
  const int l15  = lane & 15;
  const int hs   = lane >> 4;

  // A fragments: row = rt*16 + l15; lanes 0-15 K=0..15, 16-31 K=16..31 (matches B).
  v16h afr[NK];
#pragma unroll
  for (int kk = 0; kk < NK; ++kk) {
#pragma unroll
    for (int e = 0; e < 16; ++e) {
      int ka = hs * 16 + e;
      afr[kk][e] = sX[(rt * 16 + l15) * K + kk * 32 + ka];
    }
  }

  const bool fullRows = (row0 + TROWS <= N);  // block-uniform
#pragma unroll
  for (int t = 0; t < PER; ++t) {
    const int ct = cgrp * PER + t;
    v8f acc = {};
#pragma unroll
    for (int kk = 0; kk < NK; ++kk) {
      v16h bf;     // B[k][n] = W[n][k]: lanes 0-15 K=0..15, 16-31 K=16..31
#pragma unroll
      for (int e = 0; e < 16; ++e)
        bf[e] = sW[(ct * 16 + l15) * K + kk * 32 + hs * 16 + e];
      acc = __builtin_amdgcn_wmma_f32_16x16x32_f16(
          false, afr[kk], false, bf, (short)0, acc, false, false);
    }
    const int col  = ct * 16 + l15;
    const float bv = b[col];
    const int rbase = row0 + rt * 16 + hs * 8;
    if (fullRows) {                      // straight-line stores, no exec masks
#pragma unroll
      for (int r = 0; r < 8; ++r)
        Out[(size_t)(rbase + r) * M + col] = acc[r] + bv;
    } else {                             // tail block only
#pragma unroll
      for (int r = 0; r < 8; ++r) {
        int grow = rbase + r;
        if (grow < N) Out[(size_t)grow * M + col] = acc[r] + bv;
      }
    }
  }
}

// ------------------------------------------------------------ edge attention
// wave per edge: alpha = dot(q[dst], k[src]) / sqrt(d)
__global__ void edge_alpha_kernel(const float* __restrict__ Q,
                                  const float* __restrict__ Kk,
                                  const int* __restrict__ src,
                                  const int* __restrict__ dst,
                                  float* __restrict__ alpha, int E, int d) {
  int gt = blockIdx.x * blockDim.x + threadIdx.x;
  int e = gt >> 5, lane = gt & 31;
  if (e >= E) return;
  int s = src[e], t = dst[e];
  float acc = 0.0f;
  for (int c = lane; c < d; c += 32)
    acc += Q[(size_t)t * d + c] * Kk[(size_t)s * d + c];
  acc = waveSum(acc) * rsqrtf((float)d);
  if (lane == 0) alpha[e] = acc;
}

// wave per dst node: segment softmax + weighted aggregation, no atomics.
__global__ void attn_aggregate_kernel(const float* __restrict__ alpha,
                                      const float* __restrict__ V,
                                      const int* __restrict__ src,
                                      const int* __restrict__ ptr,
                                      const int* __restrict__ eid,
                                      float* __restrict__ out, int N, int d) {
  int gt = blockIdx.x * blockDim.x + threadIdx.x;
  int node = gt >> 5, lane = gt & 31;
  if (node >= N) return;
  const int beg = ptr[node], end = ptr[node + 1];

  float m = -INFINITY;                       // pass 1: segment max
  for (int j = beg + lane; j < end; j += 32) m = fmaxf(m, alpha[eid[j]]);
  m = waveMax(m);

  float s = 0.0f;                            // pass 2: sum of exp
  for (int j = beg + lane; j < end; j += 32) s += __expf(alpha[eid[j]] - m);
  s = waveSum(s);
  const float inv = 1.0f / (s + 1e-16f);

  float acc[4] = {0.0f, 0.0f, 0.0f, 0.0f};   // pass 3: weighted gather
  for (int j = beg; j < end; ++j) {
    int e  = eid[j];
    int sn = src[e];
    if (j + 1 < end)                         // gfx1250 global_prefetch_b8
      __builtin_prefetch(&V[(size_t)src[eid[j + 1]] * d], 0, 1);
    float a = __expf(alpha[e] - m) * inv;
#pragma unroll
    for (int q = 0; q < 4; ++q) {
      int c = lane + q * 32;
      if (c < d) acc[q] += a * V[(size_t)sn * d + c];
    }
  }
#pragma unroll
  for (int q = 0; q < 4; ++q) {
    int c = lane + q * 32;
    if (c < d) out[(size_t)node * d + c] = acc[q];
  }
}

// ------------------------------------------------- beta gate + LN + ReLU
__global__ void combine_kernel(const float* __restrict__ att,
                               const float* __restrict__ xr,
                               const float* __restrict__ Wbeta,
                               const float* __restrict__ lnw,
                               const float* __restrict__ lnb,
                               float* __restrict__ xout,
                               int N, int d, int do_ln) {
  int gt = blockIdx.x * blockDim.x + threadIdx.x;
  int node = gt >> 5, lane = gt & 31;
  if (node >= N) return;
  float o[4], r[4], y[4];
  float dotb = 0.0f;
#pragma unroll
  for (int j = 0; j < 4; ++j) {
    int c = lane + j * 32;
    if (c < d) {
      o[j] = att[(size_t)node * d + c];
      r[j] = xr[(size_t)node * d + c];
      dotb += o[j] * Wbeta[c] + r[j] * Wbeta[d + c] + (o[j] - r[j]) * Wbeta[2 * d + c];
    } else { o[j] = 0.0f; r[j] = 0.0f; }
  }
  dotb = waveSum(dotb);
  float beta = 1.0f / (1.0f + __expf(-dotb));
  float s1 = 0.0f;
#pragma unroll
  for (int j = 0; j < 4; ++j) {
    y[j] = beta * r[j] + (1.0f - beta) * o[j];
    s1 += (lane + j * 32 < d) ? y[j] : 0.0f;
  }
  if (do_ln) {
    s1 = waveSum(s1);
    float mu = s1 / (float)d;
    float s2 = 0.0f;
#pragma unroll
    for (int j = 0; j < 4; ++j)
      if (lane + j * 32 < d) { float t = y[j] - mu; s2 += t * t; }
    s2 = waveSum(s2);
    float inv = rsqrtf(s2 / (float)d + 1e-5f);
#pragma unroll
    for (int j = 0; j < 4; ++j) {
      int c = lane + j * 32;
      if (c < d) {
        float v = (y[j] - mu) * inv * lnw[c] + lnb[c];
        xout[(size_t)node * d + c] = v > 0.0f ? v : 0.0f;  // ReLU
      }
    }
  } else {
#pragma unroll
    for (int j = 0; j < 4; ++j) {
      int c = lane + j * 32;
      if (c < d) xout[(size_t)node * d + c] = y[j];
    }
  }
}

// ---------------------------------------------------------------- Set2Set
// one block (256 thr) per graph: gates[j] = qs@Wih[j] + bih + h@Whh[j] + bhh
__global__ void lstm_step_kernel(const float* __restrict__ qstar,
                                 const float* __restrict__ Wih,
                                 const float* __restrict__ bih,
                                 const float* __restrict__ Whh,
                                 const float* __restrict__ bhh,
                                 float* __restrict__ h, float* __restrict__ c) {
  __shared__ float sq[2 * ENC], sh[ENC], sg[4 * ENC];
  int g = blockIdx.x, tid = threadIdx.x;
  if (tid < 2 * ENC) sq[tid] = qstar[(size_t)g * 2 * ENC + tid];
  if (tid < ENC)     sh[tid] = h[(size_t)g * ENC + tid];
  __syncthreads();
  float acc = bih[tid] + bhh[tid];
  for (int k = 0; k < 2 * ENC; ++k) acc += sq[k] * Wih[(size_t)tid * 2 * ENC + k];
  for (int k = 0; k < ENC; ++k)     acc += sh[k] * Whh[(size_t)tid * ENC + k];
  sg[tid] = acc;
  __syncthreads();
  if (tid < ENC) {
    float gi = 1.0f / (1.0f + __expf(-sg[tid]));            // i
    float gf = 1.0f / (1.0f + __expf(-sg[ENC + tid]));      // f
    float gg = tanhf(sg[2 * ENC + tid]);                    // g
    float go = 1.0f / (1.0f + __expf(-sg[3 * ENC + tid]));  // o
    float cn = gf * c[(size_t)g * ENC + tid] + gi * gg;
    c[(size_t)g * ENC + tid] = cn;
    h[(size_t)g * ENC + tid] = go * tanhf(cn);
  }
}

// wave per graph: fused energy / segment-softmax / readout; batch is sorted,
// so graph g owns the contiguous node range [gptr[g], gptr[g+1]).
__global__ void s2s_step_kernel(const float* __restrict__ x,
                                const int* __restrict__ gptr,
                                const float* __restrict__ h,
                                float* __restrict__ qstar, int G) {
  int gt = blockIdx.x * blockDim.x + threadIdx.x;
  int g = gt >> 5, lane = gt & 31;
  if (g >= G) return;
  const int beg = gptr[g], end = gptr[g + 1];
  const float h0 = h[(size_t)g * ENC + lane];
  const float h1 = h[(size_t)g * ENC + lane + 32];

  float m = -INFINITY;
  for (int i = beg; i < end; ++i) {
    float e = waveSum(x[(size_t)i * ENC + lane] * h0 +
                      x[(size_t)i * ENC + lane + 32] * h1);
    m = fmaxf(m, e);
  }
  float s = 0.0f;
  for (int i = beg; i < end; ++i) {
    float e = waveSum(x[(size_t)i * ENC + lane] * h0 +
                      x[(size_t)i * ENC + lane + 32] * h1);
    s += __expf(e - m);
  }
  const float inv = 1.0f / (s + 1e-16f);
  float r0 = 0.0f, r1 = 0.0f;
  for (int i = beg; i < end; ++i) {
    float x0 = x[(size_t)i * ENC + lane];
    float x1 = x[(size_t)i * ENC + lane + 32];
    float e = waveSum(x0 * h0 + x1 * h1);
    float a = __expf(e - m) * inv;
    r0 += a * x0;
    r1 += a * x1;
  }
  qstar[(size_t)g * 2 * ENC + lane]            = h0;
  qstar[(size_t)g * 2 * ENC + lane + 32]       = h1;
  qstar[(size_t)g * 2 * ENC + ENC + lane]      = r0;
  qstar[(size_t)g * 2 * ENC + ENC + lane + 32] = r1;
}

// ----------------------------------------------------------- reparameterize
__device__ __forceinline__ unsigned pcg_hash(unsigned v) {
  v = v * 747796405u + 2891336453u;
  unsigned w = ((v >> ((v >> 28u) + 4u)) ^ v) * 277803737u;
  return (w >> 22u) ^ w;
}

__global__ void reparam_kernel(const float* __restrict__ mu,
                               const float* __restrict__ lv,
                               float* __restrict__ z, int n) {
  int i = blockIdx.x * blockDim.x + threadIdx.x;
  if (i >= n) return;
  unsigned u1 = pcg_hash(2u * (unsigned)i + 1u);
  unsigned u2 = pcg_hash(2u * (unsigned)i + 2u);
  float f1 = ((float)u1 + 0.5f) * (1.0f / 4294967296.0f);
  float f2 = ((float)u2 + 0.5f) * (1.0f / 4294967296.0f);
  float eps = sqrtf(-2.0f * logf(f1)) * __cosf(6.28318530718f * f2);
  z[i] = eps * __expf(0.5f * lv[i]) + mu[i];
}

// ---------------------------------------------------------------------------
// d_in layout assumption: top-level setup dict in insertion order
//   0:x 1:edge_attr 2:edge_index 3:batch_index, then params flattened as a
// jax pytree (dict keys sorted):
//   convs[i] (i=0..3): +0 Wbeta +1 Wk +2 Wq +3 Ws +4 Wv +5 bk +6 bq +7 bs +8 bv
//   40:ln_b 41:ln_w 42:Whh 43:Wih 44:bhh 45:bih 46:lv_W 47:lv_b 48:mu_W 49:mu_b
// ---------------------------------------------------------------------------
extern "C" void kernel_launch(void* const* d_in, const int* in_sizes, int n_in,
                              void* d_out, int out_size, void* d_ws, size_t ws_size,
                              hipStream_t stream) {
  (void)in_sizes; (void)n_in; (void)out_size; (void)ws_size;
  const float* x_in  = (const float*)d_in[0];
  const int*   eidx  = (const int*)d_in[2];
  const int*   src   = eidx;
  const int*   dst   = eidx + N_EDGES;
  const int*   batch = (const int*)d_in[3];

  auto CW = [&](int layer, int off) { return (const float*)d_in[4 + layer * 9 + off]; };
  const float* ln_b = (const float*)d_in[40];
  const float* ln_w = (const float*)d_in[41];
  const float* Whh  = (const float*)d_in[42];
  const float* Wih  = (const float*)d_in[43];
  const float* bhh  = (const float*)d_in[44];
  const float* bih  = (const float*)d_in[45];
  const float* lv_W = (const float*)d_in[46];
  const float* lv_b = (const float*)d_in[47];
  const float* mu_W = (const float*)d_in[48];
  const float* mu_b = (const float*)d_in[49];

  // ---- workspace carve-up
  float* ws = (float*)d_ws;
  size_t off = 0;
  auto allocf = [&](size_t n) { float* p = ws + off; off += n; return p; };
  float* Q   = allocf((size_t)N_NODES * EMB);
  float* Kb  = allocf((size_t)N_NODES * EMB);
  float* V   = allocf((size_t)N_NODES * EMB);
  float* S   = allocf((size_t)N_NODES * EMB);
  float* ATT = allocf((size_t)N_NODES * EMB);
  float* XC  = allocf((size_t)N_NODES * EMB);
  float* alp = allocf(N_EDGES);
  float* hg  = allocf((size_t)N_GRAPHS * ENC);
  float* cg  = allocf((size_t)N_GRAPHS * ENC);
  float* qst = allocf((size_t)N_GRAPHS * 2 * ENC);

  int* wsi = (int*)(ws + off);
  size_t ioff = 0;
  auto alloci = [&](size_t n) { int* p = wsi + ioff; ioff += n; return p; };
  int* eptr = alloci(N_NODES + 1);   // dst-CSR row pointers
  int* eid  = alloci(N_EDGES);       // edge ids bucketed by dst
  int* gptr = alloci(N_GRAPHS + 1);  // graph-CSR row pointers
  int* tmp  = alloci(N_NODES + 1);   // counts / cursor scratch

  auto fillF = [&](float* p, float v, int n) {
    fill_f32<<<(n + 255) / 256, 256, 0, stream>>>(p, v, n);
  };
  auto fillI = [&](int* p, int v, int n) {
    fill_i32<<<(n + 255) / 256, 256, 0, stream>>>(p, v, n);
  };

  // ---- build dst-CSR (once per call)
  fillI(tmp, 0, N_NODES);
  count_kernel<<<(N_EDGES + 255) / 256, 256, 0, stream>>>(dst, tmp, N_EDGES);
  scan_kernel<<<1, 1024, 0, stream>>>(tmp, eptr, N_NODES);
  copy_i32<<<(N_NODES + 255) / 256, 256, 0, stream>>>(eptr, tmp, N_NODES);
  bucket_kernel<<<(N_EDGES + 255) / 256, 256, 0, stream>>>(dst, tmp, eid, N_EDGES);

  // ---- build graph-CSR from sorted batch_index
  fillI(tmp, 0, N_GRAPHS);
  count_kernel<<<(N_NODES + 255) / 256, 256, 0, stream>>>(batch, tmp, N_NODES);
  scan_kernel<<<1, 1024, 0, stream>>>(tmp, gptr, N_GRAPHS);

  const int gemmGridN  = (N_NODES + TROWS - 1) / TROWS;   // 782
  const int edgeWaves  = (N_EDGES * 32 + 255) / 256;
  const int nodeWaves  = (N_NODES * 32 + 255) / 256;
  const int graphWaves = (N_GRAPHS * 32 + 255) / 256;

  for (int L = 0; L < NUM_LAYERS; ++L) {
    const float* xin = (L == 0) ? x_in : XC;
    const int M = (L == NUM_LAYERS - 1) ? ENC : EMB;

    if (L == 0) {                       // K=64, M=128
      gemm_bias_wmma<EMB, H_SIZE><<<gemmGridN, 256, 0, stream>>>(xin, CW(L, 2), CW(L, 6), Q,  N_NODES);
      gemm_bias_wmma<EMB, H_SIZE><<<gemmGridN, 256, 0, stream>>>(xin, CW(L, 1), CW(L, 5), Kb, N_NODES);
      gemm_bias_wmma<EMB, H_SIZE><<<gemmGridN, 256, 0, stream>>>(xin, CW(L, 4), CW(L, 8), V,  N_NODES);
      gemm_bias_wmma<EMB, H_SIZE><<<gemmGridN, 256, 0, stream>>>(xin, CW(L, 3), CW(L, 7), S,  N_NODES);
    } else if (L < NUM_LAYERS - 1) {    // K=128, M=128
      gemm_bias_wmma<EMB, EMB><<<gemmGridN, 256, 0, stream>>>(xin, CW(L, 2), CW(L, 6), Q,  N_NODES);
      gemm_bias_wmma<EMB, EMB><<<gemmGridN, 256, 0, stream>>>(xin, CW(L, 1), CW(L, 5), Kb, N_NODES);
      gemm_bias_wmma<EMB, EMB><<<gemmGridN, 256, 0, stream>>>(xin, CW(L, 4), CW(L, 8), V,  N_NODES);
      gemm_bias_wmma<EMB, EMB><<<gemmGridN, 256, 0, stream>>>(xin, CW(L, 3), CW(L, 7), S,  N_NODES);
    } else {                            // K=128, M=64
      gemm_bias_wmma<ENC, EMB><<<gemmGridN, 256, 0, stream>>>(xin, CW(L, 2), CW(L, 6), Q,  N_NODES);
      gemm_bias_wmma<ENC, EMB><<<gemmGridN, 256, 0, stream>>>(xin, CW(L, 1), CW(L, 5), Kb, N_NODES);
      gemm_bias_wmma<ENC, EMB><<<gemmGridN, 256, 0, stream>>>(xin, CW(L, 4), CW(L, 8), V,  N_NODES);
      gemm_bias_wmma<ENC, EMB><<<gemmGridN, 256, 0, stream>>>(xin, CW(L, 3), CW(L, 7), S,  N_NODES);
    }

    edge_alpha_kernel<<<edgeWaves, 256, 0, stream>>>(Q, Kb, src, dst, alp, N_EDGES, M);
    attn_aggregate_kernel<<<nodeWaves, 256, 0, stream>>>(alp, V, src, eptr, eid, ATT,
                                                         N_NODES, M);
    combine_kernel<<<nodeWaves, 256, 0, stream>>>(
        ATT, S, CW(L, 0), ln_w + L * EMB, ln_b + L * EMB, XC,
        N_NODES, M, (L < NUM_LAYERS - 1) ? 1 : 0);
  }

  // ---- Set2Set over XC [N,64]
  fillF(hg, 0.0f, N_GRAPHS * ENC);
  fillF(cg, 0.0f, N_GRAPHS * ENC);
  fillF(qst, 0.0f, N_GRAPHS * 2 * ENC);
  for (int step = 0; step < S2S_STEPS; ++step) {
    lstm_step_kernel<<<N_GRAPHS, 256, 0, stream>>>(qst, Wih, bih, Whh, bhh, hg, cg);
    s2s_step_kernel<<<graphWaves, 256, 0, stream>>>(XC, gptr, hg, qst, N_GRAPHS);
  }

  // ---- heads: z | z_mu | z_lsgms concatenated in d_out
  const int ZS = N_GRAPHS * EMB;
  float* z   = (float*)d_out;
  float* zmu = z + ZS;
  float* zlv = z + 2 * ZS;
  const int gemmGridG = (N_GRAPHS + TROWS - 1) / TROWS;   // 32
  gemm_bias_wmma<EMB, EMB><<<gemmGridG, 256, 0, stream>>>(qst, mu_W, mu_b, zmu, N_GRAPHS);
  gemm_bias_wmma<EMB, EMB><<<gemmGridG, 256, 0, stream>>>(qst, lv_W, lv_b, zlv, N_GRAPHS);
  reparam_kernel<<<(ZS + 255) / 256, 256, 0, stream>>>(zmu, zlv, z, ZS);
}
